// SelectiveSSM_9904194584644
// MI455X (gfx1250) — matrix operand order repaired and checked by reading, not compiled, not run on hardware
//
#include <hip/hip_runtime.h>
#include <math.h>
#include <stdint.h>

#define D_MODEL   1024
#define D_STATE   16
#define D_CONV    4
#define D_INNER   2048
#define DT_RANK   64
#define B_SZ      2
#define SEQ       2048
#define ROWS      (B_SZ * SEQ)      // 4096
#define XPROJ_PAD 128               // 96 padded so GEMM N is a multiple of 64

typedef __attribute__((ext_vector_type(16))) __bf16 v16bf;
typedef __attribute__((ext_vector_type(8)))  float  v8f;

struct __align__(16) U4 { unsigned a, b, c, d; };

// ---- CDNA5 async global->LDS copy (guarded; falls back to sync copy) ----
#if __has_builtin(__builtin_amdgcn_global_load_async_to_lds_b128)
#define HAVE_ASYNC_LDS 1
#else
#define HAVE_ASYNC_LDS 0
#endif

#if HAVE_ASYNC_LDS
// Builtin signature (from hipcc diagnostic): (int4 as1*, int4 as3*, imm, imm)
typedef int v4i_gcc __attribute__((vector_size(16)));
typedef v4i_gcc __attribute__((address_space(1)))* g_v4i_p;
typedef v4i_gcc __attribute__((address_space(3)))* l_v4i_p;
#endif

__device__ __forceinline__ void copy16_to_lds(const unsigned short* g, unsigned short* l) {
#if HAVE_ASYNC_LDS
    // uintptr_t round-trip: generic->AS1 keeps the VA; generic->AS3 keeps the
    // LDS offset in the low 32 bits.
    g_v4i_p gp = reinterpret_cast<g_v4i_p>(reinterpret_cast<uintptr_t>(g));
    l_v4i_p lp = reinterpret_cast<l_v4i_p>(reinterpret_cast<uintptr_t>(l));
    __builtin_amdgcn_global_load_async_to_lds_b128(gp, lp, 0, 0);
#else
    *(U4*)l = *(const U4*)g;
#endif
}

__device__ __forceinline__ void copy_wait() {
#if HAVE_ASYNC_LDS
#if __has_builtin(__builtin_amdgcn_s_wait_asynccnt)
    __builtin_amdgcn_s_wait_asynccnt(0);
#else
    asm volatile("s_wait_asynccnt 0x0" ::: "memory");
#endif
#endif
}

__device__ __forceinline__ unsigned short f2bf(float f) {
    union { float f; unsigned u; } cv; cv.f = f;
    unsigned u = cv.u;
    u += 0x7fffu + ((u >> 16) & 1u);   // round-to-nearest-even
    return (unsigned short)(u >> 16);
}

__device__ __forceinline__ float sigmoidf_(float x) {
    return 1.0f / (1.0f + __expf(-x));
}

// ---------------- elementwise f32 -> bf16 ----------------
__global__ void k_cvt(const float* __restrict__ src, unsigned short* __restrict__ dst, int n) {
    int i = blockIdx.x * blockDim.x + threadIdx.x;
    int stride = gridDim.x * blockDim.x;
    for (; i < n; i += stride) dst[i] = f2bf(src[i]);
}

// f32 src[K, scols] -> bf16 dst[drows, K] transposed; rows >= scols zero-filled.
__global__ void k_cvt_T(const float* __restrict__ src, unsigned short* __restrict__ dst,
                        int K, int scols, int drows) {
    int i = blockIdx.x * blockDim.x + threadIdx.x;
    int n = drows * K;
    int stride = gridDim.x * blockDim.x;
    for (; i < n; i += stride) {
        int row = i / K;   // output row = source column
        int k   = i % K;
        dst[i] = (row < scols) ? f2bf(src[(size_t)k * scols + row]) : (unsigned short)0;
    }
}

// f32 [rows, sstride] first `cols` columns -> bf16 [rows, cols]
__global__ void k_cvt_slice(const float* __restrict__ src, unsigned short* __restrict__ dst,
                            int rows, int sstride, int cols) {
    int i = blockIdx.x * blockDim.x + threadIdx.x;
    int n = rows * cols;
    int stride = gridDim.x * blockDim.x;
    for (; i < n; i += stride) {
        int r = i / cols, c = i % cols;
        dst[i] = f2bf(src[(size_t)r * sstride + c]);
    }
}

// ---------------- causal depthwise conv1d + SiLU ----------------
__global__ void k_conv_silu(const float* __restrict__ xz,
                            const float* __restrict__ w_conv,
                            const float* __restrict__ b_conv,
                            float* __restrict__ xssm,
                            unsigned short* __restrict__ xssm_bf) {
    int i = blockIdx.x * blockDim.x + threadIdx.x;
    const int n = ROWS * D_INNER;
    if (i >= n) return;
    const int d  = i % D_INNER;
    const int gl = i / D_INNER;
    const int l  = gl % SEQ;
    float acc = b_conv[d];
#pragma unroll
    for (int k = 0; k < D_CONV; ++k) {
        int ll = l + k - (D_CONV - 1);
        if (ll >= 0)
            acc += xz[(size_t)(gl + k - (D_CONV - 1)) * (2 * D_INNER) + d] * w_conv[d * D_CONV + k];
    }
    float s = acc * sigmoidf_(acc);
    xssm[i]    = s;
    xssm_bf[i] = f2bf(s);
}

// ---------------- selective scan ----------------
__global__ __launch_bounds__(256) void k_scan(
        const float* __restrict__ xdbc,   // [ROWS, XPROJ_PAD] (dt|B|C, padded)
        const float* __restrict__ dtf,    // [ROWS, D_INNER]   softplus(dt)
        const float* __restrict__ xssm,   // [ROWS, D_INNER]
        const float* __restrict__ xz,     // [ROWS, 2*D_INNER] (z in upper half)
        const float* __restrict__ A_log,  // [D_INNER, D_STATE]
        const float* __restrict__ Dp,     // [D_INNER]
        unsigned short* __restrict__ ygate_bf) {
    __shared__ float bc[2 * D_STATE];
    const int tid = threadIdx.x;
    const int bb  = blockIdx.x / (D_INNER / 256);
    const int d   = (blockIdx.x % (D_INNER / 256)) * 256 + tid;

    float A[D_STATE], h[D_STATE];
#pragma unroll
    for (int s = 0; s < D_STATE; ++s) {
        A[s] = -__expf(A_log[d * D_STATE + s]);
        h[s] = 0.0f;
    }
    const float Dd = Dp[d];

    for (int l = 0; l < SEQ; ++l) {
        const int row = bb * SEQ + l;
        if (tid < 2 * D_STATE) bc[tid] = xdbc[(size_t)row * XPROJ_PAD + DT_RANK + tid];
        __syncthreads();
        const float dtv = dtf[(size_t)row * D_INNER + d];
        const float xs  = xssm[(size_t)row * D_INNER + d];
        const float xdt = xs * dtv;
        float y = 0.0f;
#pragma unroll
        for (int s = 0; s < D_STATE; ++s) {
            float ab = __expf(dtv * A[s]);
            h[s] = ab * h[s] + xdt * bc[s];
            y += bc[D_STATE + s] * h[s];
        }
        const float zv  = xz[(size_t)row * (2 * D_INNER) + D_INNER + d];
        const float out = (y + xs * Dd) * (zv * sigmoidf_(zv));
        ygate_bf[(size_t)row * D_INNER + d] = f2bf(out);
        __syncthreads();
    }
}

// ---------------- bf16 WMMA GEMM (double-buffered, async LDS staging) ----------------
// C[M,N] f32 = A[M,K] bf16 @ Bt[N,K] bf16 (B pre-transposed).
// Block = 128 threads = 4 waves.  Block tile 64x64, K-step 64 (2 WMMA k-chunks).
// Both tiles contiguous [row][k] in LDS (stride 72 shorts = 144B keeps every 16B
// fragment slice aligned).  Double-buffered: async copies for step k+1 in flight
// while the 8 WMMAs of step k execute; one barrier per K-step.
#define KSTEP 64
#define LDT   72

template<int EPI>
__global__ __launch_bounds__(128) void k_gemm_bf16(
        const unsigned short* __restrict__ A,
        const unsigned short* __restrict__ Bt,
        float* __restrict__ C,
        const float* __restrict__ bias,
        int M, int N, int K) {
    __shared__ __align__(16) unsigned short lA[2][64 * LDT];
    __shared__ __align__(16) unsigned short lB[2][64 * LDT];

    const int tid  = threadIdx.x;
    const int wave = tid >> 5;
    const int lane = tid & 31;
    const int m0 = blockIdx.y * 64;
    const int n0 = blockIdx.x * 64;

    const int r  = tid >> 1;           // tile row 0..63
    const int hh = (tid & 1) * 32;     // half-row offset in shorts (64B)

    v8f acc[4] = {};

    auto stage = [&](int buf, int k0) {
        const unsigned short* ga = A  + (size_t)(m0 + r) * K + k0 + hh;
        const unsigned short* gb = Bt + (size_t)(n0 + r) * K + k0 + hh;
        unsigned short* la = &lA[buf][r * LDT + hh];
        unsigned short* lb = &lB[buf][r * LDT + hh];
#pragma unroll
        for (int c = 0; c < 4; ++c) copy16_to_lds(ga + c * 8, la + c * 8);
#pragma unroll
        for (int c = 0; c < 4; ++c) copy16_to_lds(gb + c * 8, lb + c * 8);
    };

    stage(0, 0);
    int buf = 0;
    for (int k0 = 0; k0 < K; k0 += KSTEP) {
        copy_wait();           // this wave's async copies into `buf` are done
        __syncthreads();       // ... and every other wave's too
        if (k0 + KSTEP < K) stage(buf ^ 1, k0 + KSTEP);

        // A fragment: 16-bit A 16x32 layout (ISA 7.12.2)
        //   lanes 0-15: K = {0..7, 16..23}; lanes 16-31: K = {8..15, 24..31}
        // B fragment: same K order as A (lanes 0-15: K = {0..7, 16..23}; lanes 16-31: K = {8..15, 24..31})
        const int mrow = wave * 16 + (lane & 15);
        const int klo  = (lane < 16) ? 0 : 8;
        const int kb   = (lane < 16) ? 0 : 8;
#pragma unroll
        for (int kc = 0; kc < 2; ++kc) {
            const int ko = kc * 32;
            union AF { v16bf v; U4 q[2]; } af;
            af.q[0] = *(const U4*)(&lA[buf][mrow * LDT + ko + klo]);
            af.q[1] = *(const U4*)(&lA[buf][mrow * LDT + ko + klo + 16]);
#pragma unroll
            for (int j = 0; j < 4; ++j) {
                const int ncol = j * 16 + (lane & 15);
                union BF { v16bf v; U4 q[2]; } bfg;
                bfg.q[0] = *(const U4*)(&lB[buf][ncol * LDT + ko + kb]);
                bfg.q[1] = *(const U4*)(&lB[buf][ncol * LDT + ko + kb + 16]);
                acc[j] = __builtin_amdgcn_wmma_f32_16x16x32_bf16(
                    false, af.v, false, bfg.v, (short)0, acc[j], false, false);
            }
        }
        buf ^= 1;
    }

    // epilogue: C/D layout VGPR v -> M = v (+8 for lanes 16-31), N = lane%16
    const int rbase = m0 + wave * 16 + ((lane < 16) ? 0 : 8);
    const int cl    = lane & 15;
#pragma unroll
    for (int j = 0; j < 4; ++j) {
        const int col = n0 + j * 16 + cl;
#pragma unroll
        for (int v = 0; v < 8; ++v) {
            float val = acc[j][v];
            if (EPI == 1) {
                val += bias[col];
                val = (val > 20.0f) ? val : log1pf(__expf(val));
            }
            C[(size_t)(rbase + v) * N + col] = val;
        }
    }
}

// ---------------- launcher ----------------
extern "C" void kernel_launch(void* const* d_in, const int* in_sizes, int n_in,
                              void* d_out, int out_size, void* d_ws, size_t ws_size,
                              hipStream_t stream) {
    (void)in_sizes; (void)n_in; (void)out_size; (void)ws_size;
    const float* x       = (const float*)d_in[0];
    const float* w_in    = (const float*)d_in[1];
    const float* w_conv  = (const float*)d_in[2];
    const float* b_conv  = (const float*)d_in[3];
    const float* w_xproj = (const float*)d_in[4];
    const float* w_dt    = (const float*)d_in[5];
    const float* b_dt    = (const float*)d_in[6];
    const float* A_log   = (const float*)d_in[7];
    const float* Dp      = (const float*)d_in[8];
    const float* w_out   = (const float*)d_in[9];
    float* out = (float*)d_out;

    char* ws = (char*)d_ws;
    size_t off = 0;
    auto take = [&](size_t bytes) {
        char* p = ws + off;
        off += (bytes + 255) & ~(size_t)255;
        return p;
    };
    float*          xz       = (float*)take((size_t)ROWS * 2 * D_INNER * 4);
    float*          xssm     = (float*)take((size_t)ROWS * D_INNER * 4);
    unsigned short* xssm_bf  = (unsigned short*)take((size_t)ROWS * D_INNER * 2);
    float*          xdbc     = (float*)take((size_t)ROWS * XPROJ_PAD * 4);
    unsigned short* dtlow_bf = (unsigned short*)take((size_t)ROWS * DT_RANK * 2);
    float*          dtf      = (float*)take((size_t)ROWS * D_INNER * 4);
    unsigned short* ygate_bf = (unsigned short*)take((size_t)ROWS * D_INNER * 2);
    unsigned short* x_bf     = (unsigned short*)take((size_t)ROWS * D_MODEL * 2);
    unsigned short* w_in_bf  = (unsigned short*)take((size_t)(2 * D_INNER) * D_MODEL * 2);  // [4096,1024]
    unsigned short* w_xp_bf  = (unsigned short*)take((size_t)XPROJ_PAD * D_INNER * 2);      // [128,2048]
    unsigned short* w_dt_bf  = (unsigned short*)take((size_t)D_INNER * DT_RANK * 2);        // [2048,64]
    unsigned short* w_out_bf = (unsigned short*)take((size_t)D_MODEL * D_INNER * 2);        // [1024,2048]

    const int CT = 256;
    auto grid = [&](int n) { return (n + CT - 1) / CT; };

    // ---- staging: x -> bf16; weights -> bf16 TRANSPOSED [N,K] ----
    k_cvt<<<grid(ROWS * D_MODEL), CT, 0, stream>>>(x, x_bf, ROWS * D_MODEL);
    k_cvt_T<<<grid((2 * D_INNER) * D_MODEL), CT, 0, stream>>>(
        w_in, w_in_bf, D_MODEL, 2 * D_INNER, 2 * D_INNER);
    k_cvt_T<<<grid(XPROJ_PAD * D_INNER), CT, 0, stream>>>(
        w_xproj, w_xp_bf, D_INNER, DT_RANK + 2 * D_STATE, XPROJ_PAD);
    k_cvt_T<<<grid(D_INNER * DT_RANK), CT, 0, stream>>>(
        w_dt, w_dt_bf, DT_RANK, D_INNER, D_INNER);
    k_cvt_T<<<grid(D_MODEL * D_INNER), CT, 0, stream>>>(
        w_out, w_out_bf, D_INNER, D_MODEL, D_MODEL);

    // ---- GEMM1: xz = x @ w_in ----
    k_gemm_bf16<0><<<dim3((2 * D_INNER) / 64, ROWS / 64), 128, 0, stream>>>(
        x_bf, w_in_bf, xz, nullptr, ROWS, 2 * D_INNER, D_MODEL);

    // ---- conv1d + SiLU ----
    k_conv_silu<<<grid(ROWS * D_INNER), CT, 0, stream>>>(xz, w_conv, b_conv, xssm, xssm_bf);

    // ---- GEMM2: xdbc = xssm @ w_xproj (N padded to 128) ----
    k_gemm_bf16<0><<<dim3(XPROJ_PAD / 64, ROWS / 64), 128, 0, stream>>>(
        xssm_bf, w_xp_bf, xdbc, nullptr, ROWS, XPROJ_PAD, D_INNER);

    // ---- slice dt-rank columns -> bf16 ----
    k_cvt_slice<<<grid(ROWS * DT_RANK), CT, 0, stream>>>(xdbc, dtlow_bf, ROWS, XPROJ_PAD, DT_RANK);

    // ---- GEMM3: dtf = softplus(dtlow @ w_dt + b_dt) ----
    k_gemm_bf16<1><<<dim3(D_INNER / 64, ROWS / 64), 128, 0, stream>>>(
        dtlow_bf, w_dt_bf, dtf, b_dt, ROWS, D_INNER, DT_RANK);

    // ---- selective scan + D-skip + SiLU(z) gate -> bf16 ----
    k_scan<<<B_SZ * (D_INNER / 256), 256, 0, stream>>>(
        xdbc, dtf, xssm, xz, A_log, Dp, ygate_bf);

    // ---- GEMM4: out = ygate @ w_out ----
    k_gemm_bf16<0><<<dim3(D_MODEL / 64, ROWS / 64), 128, 0, stream>>>(
        ygate_bf, w_out_bf, out, nullptr, ROWS, D_MODEL, D_INNER);
}
